// SCD_549755813972
// MI455X (gfx1250) — matrix operand order repaired and checked by reading, not compiled, not run on hardware
//
#include <hip/hip_runtime.h>

// MI455X / gfx1250, wave32. Pipeline (all on `stream`):
//   memset ws_xcon -> k_xcon (channel mean) -> k_prep (MLP + weight folding)
//   -> k_conv_wmma (folded 5x5 conv as WMMA f16 GEMM, M=16(9 valid),K=1600)
//   -> k_attn (softmax over 9 + reflect-padded 3x3 patch sum)
// Workspace use: 65536 + 8*9*128*128*4 = 4,784,128 bytes.

typedef __attribute__((ext_vector_type(16))) _Float16 v16h;
typedef __attribute__((ext_vector_type(8)))  _Float16 v8h;
typedef __attribute__((ext_vector_type(8)))  float    v8f;

#define HH 128
#define WW 128
#define CC 64
#define NPIX (HH * WW)
#define XS_STRIDE 72   // halves per (r,c) cell: 144B, breaks 128B bank period

// ------------------------------------------------------------------
// Kernel 1: per-(b,c) plane sums -> atomicAdd into xcon_sum[64]
// ------------------------------------------------------------------
__global__ __launch_bounds__(256) void k_xcon(const float* __restrict__ x,
                                              float* __restrict__ xcon_sum) {
  int c = blockIdx.x, b = blockIdx.y;
  const float4* xp = (const float4*)(x + (size_t)(b * CC + c) * NPIX);
  float s = 0.f;
  for (int i = threadIdx.x; i < NPIX / 4; i += 256) {
    float4 v = xp[i];
    s += v.x + v.y + v.z + v.w;
  }
  __shared__ float red[256];
  red[threadIdx.x] = s;
  __syncthreads();
  for (int off = 128; off > 0; off >>= 1) {
    if (threadIdx.x < off) red[threadIdx.x] += red[threadIdx.x + off];
    __syncthreads();
  }
  if (threadIdx.x == 0) atomicAdd(&xcon_sum[c], red[0]);
}

// ------------------------------------------------------------------
// Kernel 2 (1 block): consensus MLP -> tmp[64]; fold conv weights:
//   W[k,tap,ic] = sum_j w2[k,j,tap] * w1[j,ic]   (f16, rows 9..15 = 0)
//   B2[k,tap]   = sum_j w2[k,j,tap] * b1[j]
// ------------------------------------------------------------------
__global__ __launch_bounds__(256) void k_prep(
    const float* __restrict__ w1, const float* __restrict__ b1,
    const float* __restrict__ w2, const float* __restrict__ tc1,
    const float* __restrict__ tc2, const float* __restrict__ xcon_sum,
    float* __restrict__ tmp_out, float* __restrict__ B2out,
    _Float16* __restrict__ Wh) {
  __shared__ float xcon[CC];
  __shared__ float h1[16];
  int tid = threadIdx.x;
  if (tid < CC) xcon[tid] = xcon_sum[tid] * (1.0f / (8.0f * NPIX));
  __syncthreads();
  if (tid < 16) {
    float s = 0.f;
    for (int c = 0; c < CC; ++c) s += tc1[tid * CC + c] * xcon[c];
    h1[tid] = fmaxf(s, 0.f);
  }
  __syncthreads();
  if (tid < CC) {
    float s = 0.f;
    for (int o = 0; o < 16; ++o) s += tc2[tid * 16 + o] * h1[o];
    tmp_out[tid] = fmaxf(s, 0.f);
  }
  for (int i = tid; i < 9 * 25; i += 256) {
    int k = i / 25, t = i % 25;
    float s = 0.f;
    for (int j = 0; j < CC; ++j) s += w2[(k * CC + j) * 25 + t] * b1[j];
    B2out[i] = s;
  }
  for (int i = tid; i < 16 * 1600; i += 256) {
    int oc = i / 1600, rem = i % 1600;
    int tap = rem / CC, ic = rem % CC;
    float s = 0.f;
    if (oc < 9)
      for (int j = 0; j < CC; ++j)
        s += w2[(oc * CC + j) * 25 + tap] * w1[j * CC + ic];
    Wh[i] = (_Float16)s;  // K index = tap*64 + ic, contiguous per row
  }
}

// ------------------------------------------------------------------
// Kernel 3: folded 5x5 conv as WMMA GEMM. Tile = 16x16 output pixels.
// LDS im2col tile [20][20][ic] f16 (cell stride 72 halves).
// Wave w computes rows 2w, 2w+1 (16 pixel cols on lanes). 50 K-steps of 32.
// ------------------------------------------------------------------
__global__ __launch_bounds__(256) void k_conv_wmma(
    const float* __restrict__ x, const _Float16* __restrict__ Wh,
    const float* __restrict__ B2, float* __restrict__ wt) {
  __shared__ __align__(16) _Float16 xs[20 * 20 * XS_STRIDE];
  __shared__ float sB2[9 * 25];
  int b = blockIdx.z;
  int r0 = blockIdx.y * 16;
  int c0 = blockIdx.x * 16;
  int tid = threadIdx.x;

  for (int i = tid; i < 9 * 25; i += 256) sB2[i] = B2[i];

  // Stage zero-padded halo, transposed to [pos][ic], f32 -> f16.
  for (int ic = 0; ic < CC; ++ic) {
    const float* xp = x + (size_t)(b * CC + ic) * NPIX;
    if (ic + 1 < CC) __builtin_prefetch(xp + NPIX, 0, 0);  // global_prefetch_b8
    for (int p = tid; p < 400; p += 256) {
      int r = p / 20, c = p % 20;
      int gh = r0 - 2 + r, gw = c0 - 2 + c;
      float v = (gh >= 0 && gh < HH && gw >= 0 && gw < WW) ? xp[gh * WW + gw] : 0.f;
      xs[p * XS_STRIDE + ic] = (_Float16)v;
    }
  }
  __syncthreads();

  int wave = tid >> 5, lane = tid & 31;
  int lhalf = lane >> 4;   // 0 for lanes 0-15, 1 for lanes 16-31
  int nidx = lane & 15;    // B column (pixel) / A row (m)
  int rrA = wave * 2, rrB = rrA + 1;
  v8f acc0 = {}, acc1 = {};

  for (int s = 0; s < 50; ++s) {
    int tap = s >> 1;
    int dy = tap / 5, dx = tap - dy * 5;
    int icb = ((s & 1) << 5) + (lhalf << 3);  // B: ic 8*lhalf+0..7 and +16..23 per lane
    // A fragment: lane row m=nidx; halves 0-7 = K 32s+8*lhalf+0..7,
    // halves 8-15 = K 32s+8*lhalf+16..23 (ISA 16-bit A 16x32 layout).
    const _Float16* ap = Wh + nidx * 1600 + 32 * s + 8 * lhalf;
    v8h a0 = *(const v8h*)ap;
    v8h a1 = *(const v8h*)(ap + 16);
    v16h a = __builtin_shufflevector(a0, a1, 0, 1, 2, 3, 4, 5, 6, 7,
                                     8, 9, 10, 11, 12, 13, 14, 15);
    {
      const _Float16* bp =
          &xs[((rrA + dy) * 20 + (nidx + dx)) * XS_STRIDE + icb];
      v8h p0 = *(const v8h*)bp;
      v8h p1 = *(const v8h*)(bp + 16);
      v16h bm = __builtin_shufflevector(p0, p1, 0, 1, 2, 3, 4, 5, 6, 7,
                                        8, 9, 10, 11, 12, 13, 14, 15);
      acc0 = __builtin_amdgcn_wmma_f32_16x16x32_f16(false, a, false, bm,
                                                    (short)0, acc0, false, false);
    }
    {
      const _Float16* bp =
          &xs[((rrB + dy) * 20 + (nidx + dx)) * XS_STRIDE + icb];
      v8h p0 = *(const v8h*)bp;
      v8h p1 = *(const v8h*)(bp + 16);
      v16h bm = __builtin_shufflevector(p0, p1, 0, 1, 2, 3, 4, 5, 6, 7,
                                        8, 9, 10, 11, 12, 13, 14, 15);
      acc1 = __builtin_amdgcn_wmma_f32_16x16x32_f16(false, a, false, bm,
                                                    (short)0, acc1, false, false);
    }
  }

  // D layout: lane col n = lane&15, row m = vgpr + 8*lhalf. Add per-tap bias
  // (bias of y only contributes where the conv tap lands in-bounds).
  for (int g = 0; g < 2; ++g) {
    v8f acc = g ? acc1 : acc0;
    int hh = r0 + (g ? rrB : rrA);
    int wcol = c0 + nidx;
    for (int v = 0; v < 8; ++v) {
      int m = v + 8 * lhalf;
      if (m < 9) {
        float bias = 0.f;
        for (int t = 0; t < 25; ++t) {
          int dy = t / 5, dx = t % 5;
          int sh = hh + dy - 2, sw = wcol + dx - 2;
          if (sh >= 0 && sh < HH && sw >= 0 && sw < WW) bias += sB2[m * 25 + t];
        }
        wt[(((size_t)b * 9 + m) * HH + hh) * WW + wcol] = acc[v] + bias;
      }
    }
  }
}

// ------------------------------------------------------------------
// Kernel 4: softmax over 9 logits (wt[k]*tmp[c]) + reflect-pad patch sum
// ------------------------------------------------------------------
__device__ __forceinline__ int refl(int i) {
  i = (i < 0) ? -i : i;
  return (i > HH - 1) ? (2 * (HH - 1) - i) : i;
}

__global__ __launch_bounds__(256) void k_attn(
    const float* __restrict__ x, const float* __restrict__ wtbuf,
    const float* __restrict__ tmpv, float* __restrict__ out) {
  __shared__ float swt[9 * 256];
  __shared__ float sx[18 * 19];
  int b = blockIdx.z;
  int r0 = blockIdx.y * 16;
  int c0 = blockIdx.x * 16;
  int tid = threadIdx.x;
  int rr = tid >> 4, cc = tid & 15;

  for (int i = tid; i < 9 * 256; i += 256) {
    int k = i >> 8, p = i & 255;
    int r = p >> 4, c = p & 15;
    swt[i] = wtbuf[(((size_t)b * 9 + k) * HH + (r0 + r)) * WW + (c0 + c)];
  }
  for (int ch = 0; ch < CC; ++ch) {
    __syncthreads();  // also covers swt fill on first iteration
    const float* xp = x + (size_t)(b * CC + ch) * NPIX;
    for (int i = tid; i < 18 * 18; i += 256) {
      int r = i / 18, c = i % 18;
      int gh = refl(r0 - 1 + r), gw = refl(c0 - 1 + c);
      sx[r * 19 + c] = xp[gh * WW + gw];
    }
    __syncthreads();
    float t = tmpv[ch];
    float e[9];
    float mx = -3.4e38f;
    for (int k = 0; k < 9; ++k) {
      e[k] = swt[k * 256 + tid] * t;
      mx = fmaxf(mx, e[k]);
    }
    float se = 0.f;
    for (int k = 0; k < 9; ++k) {
      e[k] = __expf(e[k] - mx);
      se += e[k];
    }
    float o = 0.f;
    for (int k = 0; k < 9; ++k) {
      int ti = k / 3, tj = k % 3;
      o += e[k] * sx[(rr + ti) * 19 + (cc + tj)];
    }
    out[((size_t)(b * CC + ch) * HH + (r0 + rr)) * WW + (c0 + cc)] = o / se;
  }
}

// ------------------------------------------------------------------
extern "C" void kernel_launch(void* const* d_in, const int* in_sizes, int n_in,
                              void* d_out, int out_size, void* d_ws, size_t ws_size,
                              hipStream_t stream) {
  const float* x   = (const float*)d_in[0];
  const float* w1  = (const float*)d_in[1];
  const float* b1  = (const float*)d_in[2];
  const float* w2  = (const float*)d_in[3];
  const float* tc1 = (const float*)d_in[4];
  const float* tc2 = (const float*)d_in[5];
  float* out = (float*)d_out;

  char* ws = (char*)d_ws;
  float*    xcon_sum = (float*)(ws + 0);       // 64 f32
  float*    tmpv     = (float*)(ws + 256);     // 64 f32
  float*    B2       = (float*)(ws + 512);     // 225 f32
  _Float16* Wh       = (_Float16*)(ws + 2048); // 16*1600 f16 = 51200 B
  float*    wtbuf    = (float*)(ws + 65536);   // 8*9*128*128 f32 = 4.5 MiB

  hipMemsetAsync(xcon_sum, 0, 64 * sizeof(float), stream);
  k_xcon<<<dim3(CC, 8), 256, 0, stream>>>(x, xcon_sum);
  k_prep<<<1, 256, 0, stream>>>(w1, b1, w2, tc1, tc2, xcon_sum, tmpv, B2, Wh);
  k_conv_wmma<<<dim3(8, 8, 8), 256, 0, stream>>>(x, Wh, B2, wtbuf);
  k_attn<<<dim3(8, 8, 8), 256, 0, stream>>>(x, wtbuf, tmpv, out);
}
